// FourierEncoding_77455440216333
// MI455X (gfx1250) — matrix operand order repaired and checked by reading, not compiled, not run on hardware
//
#include <hip/hip_runtime.h>

// CDNA5 / gfx1250, wave32.
typedef __attribute__((ext_vector_type(16))) _Float16 v16h;
typedef __attribute__((ext_vector_type(2)))  __fp16   pk2;   // cvt_pkrtz result type
typedef __attribute__((ext_vector_type(8)))  float    v8f;

#define TWO_PI 6.28318530717958647692f

// out[i] = W[0] + b + sum_{k=1..128} W[k] * cos(k*pi*e[i])
//
// Per wave: one 16-element tile per loop iteration.
//   A (16x32 f16) = W[1..128] chunk, identical rows (hoisted).
//   B (32x16 f16) = cos features; column n = element n (lane&15),
//                   lane half h = K half (K = 16h..16h+15 of the 32-chunk).
//   D = A x B + C accumulated over 4 chunks -> every D row = per-element dot.
// Features via Chebyshev: ~5 v_cos seeds + stride-1 f32 runs (tiles 0,1),
// then stride-32 *packed f16* recurrence (tiles 2,3) -> v_pk_fma_f16 emits
// two features per instruction already in B layout.
__global__ __launch_bounds__(256) void fourier_wmma_kernel(
    const float* __restrict__ E, const float* __restrict__ W,
    const float* __restrict__ Bias, float* __restrict__ out,
    int n, int ntiles)
{
  const int lane = threadIdx.x & 31;
  const int h    = lane >> 4;   // K half within a 32-chunk
  const int col  = lane & 15;   // element (B column) within the tile
  const int wave   = blockIdx.x * (blockDim.x >> 5) + (threadIdx.x >> 5);
  const int nwaves = gridDim.x * (blockDim.x >> 5);

  // ---- hoisted A operand: w replicated rows, 16x32 A in contiguous K order (K = 16h+s) ----
  // lanes 0-15: VGPR v -> K=2v,2v+1 (+16 for h=1), contiguous to match B
  v16h a[4];
#pragma unroll
  for (int t = 0; t < 4; ++t) {
#pragma unroll
    for (int v = 0; v < 8; ++v) {
      const int kap = 2 * v + 16 * h;
      const int k0  = 32 * t + kap + 1;            // w index 1..128
      pk2 p = __builtin_amdgcn_cvt_pkrtz(W[k0], W[k0 + 1]);
      a[t][2 * v]     = (_Float16)p[0];
      a[t][2 * v + 1] = (_Float16)p[1];
    }
  }
  const float w0b = W[0] + Bias[0];               // ones-column + bias

  // cos(k*pi*x) = cos(2*pi * (k*x/2)) : feed v_cos_f32 with fract(k*x/2)*2pi
  const float kf1  = (float)(16 * h + 1);
  const float kf2  = (float)(16 * h + 2);
  const float kf33 = (float)(16 * h + 33);
  const float kf34 = (float)(16 * h + 34);

  for (int tile = wave; tile < ntiles; tile += nwaves) {
    const int base = tile * 16;
    int idx = base + col;
    if (idx >= n) idx = n - 1;                    // clamp: keep EXEC all-1s
    const float x  = E[idx];
    const float hx = 0.5f * x;

    // seeds: cos(theta) and first two members of each stride-1 run
    const float cth = __builtin_amdgcn_cosf(TWO_PI * __builtin_amdgcn_fractf(hx));
    const float twoc = cth + cth;

    float r1[16], r2[16];   // k = 16h+1..16h+16  and  k = 16h+33..16h+48
    r1[0] = __builtin_amdgcn_cosf(TWO_PI * __builtin_amdgcn_fractf(kf1 * hx));
    r1[1] = __builtin_amdgcn_cosf(TWO_PI * __builtin_amdgcn_fractf(kf2 * hx));
#pragma unroll
    for (int j = 2; j < 16; ++j)
      r1[j] = __builtin_fmaf(twoc, r1[j - 1], -r1[j - 2]);
    r2[0] = __builtin_amdgcn_cosf(TWO_PI * __builtin_amdgcn_fractf(kf33 * hx));
    r2[1] = __builtin_amdgcn_cosf(TWO_PI * __builtin_amdgcn_fractf(kf34 * hx));
#pragma unroll
    for (int j = 2; j < 16; ++j)
      r2[j] = __builtin_fmaf(twoc, r2[j - 1], -r2[j - 2]);

    // cos(32*theta) by 5 doublings: cos(2p) = 2cos(p)^2 - 1
    float d32 = cth;
#pragma unroll
    for (int i = 0; i < 5; ++i)
      d32 = __builtin_fmaf(d32 + d32, d32, -1.0f);

    // pack tiles 0,1 of B (seed runs) to f16
    v16h b0, b1;
#pragma unroll
    for (int v = 0; v < 8; ++v) {
      pk2 p = __builtin_amdgcn_cvt_pkrtz(r1[2 * v], r1[2 * v + 1]);
      b0[2 * v]     = (_Float16)p[0];
      b0[2 * v + 1] = (_Float16)p[1];
      pk2 q = __builtin_amdgcn_cvt_pkrtz(r2[2 * v], r2[2 * v + 1]);
      b1[2 * v]     = (_Float16)q[0];
      b1[2 * v + 1] = (_Float16)q[1];
    }

    // tiles 2,3: stride-32 Chebyshev step in packed f16
    //   X_{t+1} = 2cos(32*theta) * X_t - X_{t-1}   (v_pk_fma_f16)
    const _Float16 dh = (_Float16)(d32 + d32);
    v16h dv;
#pragma unroll
    for (int i = 0; i < 16; ++i) dv[i] = dh;
    v16h b2 = dv * b1 - b0;
    v16h b3 = dv * b2 - b1;

    // 128-length dot on the matrix pipe (co-executes with VALU above)
    v8f acc = {};
    acc = __builtin_amdgcn_wmma_f32_16x16x32_f16(false, a[0], false, b0,
                                                 (short)0, acc, false, false);
    acc = __builtin_amdgcn_wmma_f32_16x16x32_f16(false, a[1], false, b1,
                                                 (short)0, acc, false, false);
    acc = __builtin_amdgcn_wmma_f32_16x16x32_f16(false, a[2], false, b2,
                                                 (short)0, acc, false, false);
    acc = __builtin_amdgcn_wmma_f32_16x16x32_f16(false, a[3], false, b3,
                                                 (short)0, acc, false, false);

    // D row 0 (VGPR 0, lanes 0-15) = result for element col
    const float res = acc[0] + w0b;
    if (lane < 16 && base + col < n) out[base + col] = res;
  }
}

extern "C" void kernel_launch(void* const* d_in, const int* in_sizes, int n_in,
                              void* d_out, int out_size, void* d_ws, size_t ws_size,
                              hipStream_t stream) {
  const float* e = (const float*)d_in[0];
  const float* W = (const float*)d_in[1];   // [1, 129]
  const float* b = (const float*)d_in[2];   // [1]
  float* out = (float*)d_out;               // [N]
  const int n = in_sizes[0];
  const int ntiles = (n + 15) / 16;
  // 1024 blocks x 8 waves = 8192 waves grid-striding 125K tiles (~15 each):
  // amortizes the hoisted A-operand (W) setup.
  fourier_wmma_kernel<<<1024, 256, 0, stream>>>(e, W, b, out, n, ntiles);
}
